// Attention3_11751030522302
// MI455X (gfx1250) — matrix operand order repaired and checked by reading, not compiled, not run on hardware
//
#include <hip/hip_runtime.h>

typedef __attribute__((ext_vector_type(16))) _Float16 v16h;
typedef __attribute__((ext_vector_type(8)))  _Float16 v8h;
typedef __attribute__((ext_vector_type(8)))  float    v8f;

union AFrag { v16h v; unsigned int u[8]; _Float16 f[16]; };
union BFrag { v16h v; v8h h[2]; };

#define B_SZ 8
#define NKV  1024

// ---------- weight transpose + f32->f16 : W[k][n] -> WT[n][k] ----------
__global__ void wtrans_kernel(const float* __restrict__ W, _Float16* __restrict__ WT,
                              int K, int N, int total) {
  int idx = blockIdx.x * blockDim.x + threadIdx.x;
  if (idx >= total) return;
  int k = idx / N, n = idx % N;
  WT[(size_t)n * K + k] = (_Float16)W[idx];
}

// ---------- V transpose: kvh [B][1024][128] (v at +64) -> VT [B][64][1024] ----------
__global__ void vtrans_kernel(const _Float16* __restrict__ KV, _Float16* __restrict__ VT) {
  int idx = blockIdx.x * blockDim.x + threadIdx.x;  // B*1024*64
  if (idx >= B_SZ * NKV * 64) return;
  int d  = idx & 63;
  int kv = (idx >> 6) & (NKV - 1);
  int b  = idx >> 16;
  VT[((size_t)b * 64 + d) * NKV + kv] = KV[((size_t)b * NKV + kv) * 128 + 64 + d];
}

// ---------- fused LayerNorm + GEMM:  Y(f16)[M][N] = LN(X[M][K]) @ W[K][N] ----------
template<int K, int N>
__global__ __launch_bounds__(128) void ln_gemm_kernel(
    const float* __restrict__ X, const float* __restrict__ G, const float* __restrict__ Bb,
    const _Float16* __restrict__ WT, _Float16* __restrict__ Y)
{
  __shared__ _Float16 As[16 * K];
  __shared__ float ps[16][8], ps2[16][8];
  const int tid = threadIdx.x;
  const int r = tid >> 3, sub = tid & 7;
  const size_t rowBase = (size_t)blockIdx.x * 16;
  const float* xr = X + (rowBase + r) * K;

  float s = 0.f, s2 = 0.f;
  for (int k = sub; k < K; k += 8) { float xv = xr[k]; s += xv; s2 += xv * xv; }
  ps[r][sub] = s; ps2[r][sub] = s2;
  __syncthreads();
  float mean = 0.f, m2 = 0.f;
  for (int j = 0; j < 8; ++j) { mean += ps[r][j]; m2 += ps2[r][j]; }
  mean *= (1.0f / K);
  float var = m2 * (1.0f / K) - mean * mean;
  float inv = rsqrtf(var + 1e-5f);
  for (int k = sub; k < K; k += 8)
    As[r * K + k] = (_Float16)((xr[k] - mean) * inv * G[k] + Bb[k]);
  __syncthreads();

  const int lane = tid & 31, wave = tid >> 5;
  const int mA = lane & 15;
  const int base8 = (lane >> 4) * 8;
  const unsigned int* Arow = (const unsigned int*)(As + (size_t)mA * K);

  for (int nt = wave; nt < N / 16; nt += 4) {
    const int n = nt * 16 + (lane & 15);
    const _Float16* Brow = WT + (size_t)n * K + (lane >> 4) * 8;
    v8f acc = {};
    for (int kb = 0; kb < K; kb += 32) {
      AFrag a; BFrag bf;
      #pragma unroll
      for (int j = 0; j < 8; ++j) {
        int hh = kb + base8 + ((j < 4) ? 2 * j : 2 * j + 8);
        a.u[j] = Arow[hh >> 1];
      }
      bf.h[0] = *(const v8h*)(Brow + kb);
      bf.h[1] = *(const v8h*)(Brow + kb + 16);
      acc = __builtin_amdgcn_wmma_f32_16x16x32_f16(false, a.v, false, bf.v,
                                                   (short)0, acc, false, false);
    }
    #pragma unroll
    for (int v = 0; v < 8; ++v) {
      int m = v + (lane >> 4) * 8;
      Y[(rowBase + m) * N + n] = (_Float16)acc[v];
    }
  }
}

// ---------- attention: 4 waves per 16-query tile, full softmax in LDS ----------
__global__ __launch_bounds__(128) void attn_kernel(
    const _Float16* __restrict__ Q, const _Float16* __restrict__ KV,
    const _Float16* __restrict__ VT, _Float16* __restrict__ OUTS,
    int rowsPerHead, int qStride, int mode)
{
  __shared__ float    S[16][NKV];     // 64 KB scores
  __shared__ _Float16 P[16][NKV];     // 32 KB normalized probs (f16)
  __shared__ float pmax[16][8];       // per-row partial max (8 segments)
  __shared__ float psum[16][8];       // per-row partial sum
  const int tid = threadIdx.x;
  const int lane = tid & 31, wave = tid >> 5;
  const int tile = blockIdx.x, head = blockIdx.y, b = blockIdx.z;
  const int mA = lane & 15;
  const int base8 = (lane >> 4) * 8;

  // Q fragments (head dim 64 -> 2 k-chunks); identical per wave, lives in regs
  const int qrow = tile * 16 + mA;
  const unsigned int* Qrow =
      (const unsigned int*)(Q + ((size_t)b * rowsPerHead + qrow) * qStride + head * 64);
  AFrag qa[2];
  #pragma unroll
  for (int c = 0; c < 2; ++c) {
    #pragma unroll
    for (int j = 0; j < 8; ++j) {
      int hh = c * 32 + base8 + ((j < 4) ? 2 * j : 2 * j + 8);
      qa[c].u[j] = Qrow[hh >> 1];
    }
  }

  // S = scale * Q @ K^T : wave w computes kv tile pairs {2w,2w+1}, {2w+8,2w+9}, ...
  for (int nt = wave * 2; nt < NKV / 16; nt += 8) {
    const int n0 = nt * 16 + (lane & 15);
    const int n1 = n0 + 16;
    const _Float16* Krow0 = KV + ((size_t)b * NKV + n0) * 128 + (lane >> 4) * 8;
    const _Float16* Krow1 = KV + ((size_t)b * NKV + n1) * 128 + (lane >> 4) * 8;
    v8f acc0 = {}, acc1 = {};
    #pragma unroll
    for (int c = 0; c < 2; ++c) {
      BFrag bf0, bf1;
      bf0.h[0] = *(const v8h*)(Krow0 + c * 32);
      bf0.h[1] = *(const v8h*)(Krow0 + c * 32 + 16);
      bf1.h[0] = *(const v8h*)(Krow1 + c * 32);
      bf1.h[1] = *(const v8h*)(Krow1 + c * 32 + 16);
      acc0 = __builtin_amdgcn_wmma_f32_16x16x32_f16(false, qa[c].v, false, bf0.v,
                                                    (short)0, acc0, false, false);
      acc1 = __builtin_amdgcn_wmma_f32_16x16x32_f16(false, qa[c].v, false, bf1.v,
                                                    (short)0, acc1, false, false);
    }
    #pragma unroll
    for (int v = 0; v < 8; ++v) {
      S[v + (lane >> 4) * 8][n0] = acc0[v] * 0.125f;   // 64^-0.5
      S[v + (lane >> 4) * 8][n1] = acc1[v] * 0.125f;
    }
  }
  __syncthreads();

  // softmax: 8 lanes per row, 128-column segments, cross-wave reduce via LDS
  const int r   = tid & 15;
  const int seg = tid >> 4;                 // 0..7
  const int cs  = seg * (NKV / 8);
  float mx = -3.0e38f;
  for (int c = 0; c < NKV / 8; ++c) mx = fmaxf(mx, S[r][cs + c]);
  pmax[r][seg] = mx;
  __syncthreads();
  float rmx = pmax[r][0];
  #pragma unroll
  for (int j = 1; j < 8; ++j) rmx = fmaxf(rmx, pmax[r][j]);
  float sum = 0.f;
  for (int c = 0; c < NKV / 8; ++c) {
    float e = __expf(S[r][cs + c] - rmx);
    S[r][cs + c] = e;
    sum += e;
  }
  psum[r][seg] = sum;
  __syncthreads();

  // single cooperative normalize + f32->f16 convert pass (done once, not per frag)
  {
    const int pr  = tid >> 3;               // 0..15
    const int pc0 = (tid & 7) * (NKV / 8);
    float prs = 0.f;
    #pragma unroll
    for (int j = 0; j < 8; ++j) prs += psum[pr][j];
    const float pinv = 1.0f / prs;
    for (int c = 0; c < NKV / 8; ++c)
      P[pr][pc0 + c] = (_Float16)(S[pr][pc0 + c] * pinv);
  }
  __syncthreads();

  // O = P @ V : wave w owns head-dim tile dt = w
  {
    const int dt = wave;
    const int n = dt * 16 + (lane & 15);
    const _Float16* Vrow = VT + ((size_t)b * 64 + n) * NKV + (lane >> 4) * 8;
    const unsigned int* Prow = (const unsigned int*)(&P[mA][0]);
    v8f acc = {};
    for (int kb = 0; kb < NKV; kb += 32) {
      AFrag pa; BFrag bf;
      #pragma unroll
      for (int j = 0; j < 8; ++j) {
        int hh = kb + base8 + ((j < 4) ? 2 * j : 2 * j + 8);
        pa.u[j] = Prow[hh >> 1];
      }
      bf.h[0] = *(const v8h*)(Vrow + kb);
      bf.h[1] = *(const v8h*)(Vrow + kb + 16);
      acc = __builtin_amdgcn_wmma_f32_16x16x32_f16(false, pa.v, false, bf.v,
                                                   (short)0, acc, false, false);
    }
    #pragma unroll
    for (int v = 0; v < 8; ++v) {
      int m = v + (lane >> 4) * 8;
      int qi = tile * 16 + m;
      int row, col;
      if (mode == 0) {  // o0: (B,2,4096,64) -> (B,2,1024,256) -> cols [g*320, g*320+256)
        row = qi >> 2;
        col = head * 320 + (qi & 3) * 64 + dt * 16 + (lane & 15);
      } else {          // o2: (B,8,256,64) -> (B,2,1024,64) -> cols [g*320+256, g*320+320)
        row = (head & 3) * 256 + qi;
        col = (head >> 2) * 320 + 256 + dt * 16 + (lane & 15);
      }
      OUTS[((size_t)b * NKV + row) * 640 + col] = (_Float16)acc[v];
    }
  }
}

// ---------- final GEMM: out(f32)[8192][512] = outs(f16)[8192][640] @ Wout ----------
__global__ __launch_bounds__(128) void out_gemm_kernel(
    const _Float16* __restrict__ A, const _Float16* __restrict__ WT,
    float* __restrict__ Out)
{
  const int tid = threadIdx.x, lane = tid & 31, wave = tid >> 5;
  const size_t rowBase = (size_t)blockIdx.x * 16;
  const int base8 = (lane >> 4) * 8;
  const unsigned int* Arow = (const unsigned int*)(A + (rowBase + (lane & 15)) * 640);

  for (int nt = wave; nt < 32; nt += 4) {
    const int n = nt * 16 + (lane & 15);
    const _Float16* Brow = WT + (size_t)n * 640 + (lane >> 4) * 8;
    v8f acc = {};
    for (int kb = 0; kb < 640; kb += 32) {
      AFrag a; BFrag bf;
      #pragma unroll
      for (int j = 0; j < 8; ++j) {
        int hh = kb + base8 + ((j < 4) ? 2 * j : 2 * j + 8);
        a.u[j] = Arow[hh >> 1];
      }
      bf.h[0] = *(const v8h*)(Brow + kb);
      bf.h[1] = *(const v8h*)(Brow + kb + 16);
      acc = __builtin_amdgcn_wmma_f32_16x16x32_f16(false, a.v, false, bf.v,
                                                   (short)0, acc, false, false);
    }
    #pragma unroll
    for (int v = 0; v < 8; ++v) {
      size_t m = rowBase + v + (lane >> 4) * 8;
      Out[m * 512 + n] = acc[v];
    }
  }
}

extern "C" void kernel_launch(void* const* d_in, const int* in_sizes, int n_in,
                              void* d_out, int out_size, void* d_ws, size_t ws_size,
                              hipStream_t stream) {
  const float* x0   = (const float*)d_in[0];
  const float* x1   = (const float*)d_in[1];
  const float* x2   = (const float*)d_in[2];
  const float* g0   = (const float*)d_in[3];
  const float* b0   = (const float*)d_in[4];
  const float* g1   = (const float*)d_in[5];
  const float* b1   = (const float*)d_in[6];
  const float* g2   = (const float*)d_in[7];
  const float* b2   = (const float*)d_in[8];
  const float* Wq0  = (const float*)d_in[9];
  const float* Wkv  = (const float*)d_in[10];
  const float* Wq2  = (const float*)d_in[11];
  const float* Wout = (const float*)d_in[12];
  float* out = (float*)d_out;

  _Float16* p = (_Float16*)d_ws;
  _Float16* wq0t  = p; p += 256 * 128;
  _Float16* wkvt  = p; p += 512 * 128;
  _Float16* wq2t  = p; p += 1024 * 512;
  _Float16* woutt = p; p += 640 * 512;
  _Float16* q0h   = p; p += (size_t)B_SZ * 4096 * 128;
  _Float16* kvh   = p; p += (size_t)B_SZ * 1024 * 128;
  _Float16* q2h   = p; p += (size_t)B_SZ * 256 * 512;
  _Float16* vth   = p; p += (size_t)B_SZ * 64 * 1024;
  _Float16* outsh = p; p += (size_t)B_SZ * 1024 * 640;
  (void)ws_size; (void)in_sizes; (void)n_in; (void)out_size;

  auto cdiv = [](int a, int b) { return (a + b - 1) / b; };
  wtrans_kernel<<<cdiv(256 * 128, 256), 256, 0, stream>>>(Wq0,  wq0t,  256,  128, 256 * 128);
  wtrans_kernel<<<cdiv(512 * 128, 256), 256, 0, stream>>>(Wkv,  wkvt,  512,  128, 512 * 128);
  wtrans_kernel<<<cdiv(1024 * 512, 256), 256, 0, stream>>>(Wq2,  wq2t, 1024,  512, 1024 * 512);
  wtrans_kernel<<<cdiv(640 * 512, 256), 256, 0, stream>>>(Wout, woutt, 640,  512, 640 * 512);

  ln_gemm_kernel<256, 128><<<(B_SZ * 4096) / 16, 128, 0, stream>>>(x0, g0, b0, wq0t, q0h);
  ln_gemm_kernel<512, 128><<<(B_SZ * 1024) / 16, 128, 0, stream>>>(x1, g1, b1, wkvt, kvh);
  ln_gemm_kernel<1024, 512><<<(B_SZ * 256) / 16, 128, 0, stream>>>(x2, g2, b2, wq2t, q2h);

  vtrans_kernel<<<cdiv(B_SZ * NKV * 64, 256), 256, 0, stream>>>(kvh, vth);

  attn_kernel<<<dim3(4096 / 16, 2, B_SZ), 128, 0, stream>>>(q0h, kvh, vth, outsh, 4096, 128, 0);
  attn_kernel<<<dim3(256 / 16, 8, B_SZ), 128, 0, stream>>>(q2h, kvh, vth, outsh, 256, 512, 1);

  out_gemm_kernel<<<(B_SZ * 1024) / 16, 128, 0, stream>>>(outsh, woutt, out);
}
